// tDPMPN_78228534329587
// MI455X (gfx1250) — matrix operand order repaired and checked by reading, not compiled, not run on hardware
//
#include <hip/hip_runtime.h>
#include <hip/hip_bf16.h>

typedef _Float16 half16 __attribute__((ext_vector_type(16)));
typedef float    floatx8 __attribute__((ext_vector_type(8)));

// ---- problem dims (match reference) ----
#define DD   128
#define IDD  512          // 4*D
#define ODD  256          // 2*D
#define TILE_E 32         // edges per block (2 M-tiles of 16)
#define NWAVES 8          // 256 threads

// ---------------- K0: weight swizzle (f32 -> f16 WMMA B-fragment order) + zero init ----------------
// B-fragment layout (32x16 f16 KxN): lane group g=lane/16 covers K = g*8 + 0..7 and 16 + g*8 + 0..7,
// column n = ntile*16 + lane%16, lane's 16 halves = ascending K pairs.
// Swizzled index: ((kt*(OD/16)+nt)*32 + lane)*16 + e  -> each lane reads 32B contiguously.
__global__ void k0_prep(const float* __restrict__ Wl, const float* __restrict__ Wr,
                        const float* __restrict__ Wc,
                        _Float16* __restrict__ Wl_s, _Float16* __restrict__ Wr_s,
                        _Float16* __restrict__ Wc_s,
                        unsigned* __restrict__ maxkey, float* __restrict__ sumexp,
                        float* __restrict__ out, unsigned* __restrict__ hist1,
                        unsigned* __restrict__ hist2, int N, int B)
{
    const int stride = gridDim.x * blockDim.x;
    const int tid0 = blockIdx.x * blockDim.x + threadIdx.x;

    // Wl / Wr : ID x OD
    for (int idx = tid0; idx < IDD * ODD; idx += stride) {
        int tile = idx >> 9;          // (kt*16 + nt)
        int lane = (idx >> 4) & 31;
        int e    = idx & 15;
        int kt = tile >> 4, nt = tile & 15;
        int g = lane >> 4;
        int n = nt * 16 + (lane & 15);
        int v = e >> 1, hb = e & 1;
        int kloc = ((v & 4) << 2) + ((v & 3) << 1) + hb + (g << 3); // B layout: g*8
        int k = kt * 32 + kloc;
        Wl_s[idx] = (_Float16)Wl[k * ODD + n];
        Wr_s[idx] = (_Float16)Wr[k * ODD + n];
    }
    // Wc : OD x OD
    for (int idx = tid0; idx < ODD * ODD; idx += stride) {
        int tile = idx >> 9;
        int lane = (idx >> 4) & 31;
        int e    = idx & 15;
        int kt = tile >> 4, nt = tile & 15;
        int g = lane >> 4;
        int n = nt * 16 + (lane & 15);
        int v = e >> 1, hb = e & 1;
        int kloc = ((v & 4) << 2) + ((v & 3) << 1) + hb + (g << 3);
        int k = kt * 32 + kloc;
        Wc_s[idx] = (_Float16)Wc[k * ODD + n];
    }
    for (int i = tid0; i < N; i += stride) { maxkey[i] = 0u; sumexp[i] = 0.0f; out[i] = 0.0f; }
    for (int i = tid0; i < B * 256; i += stride) { hist1[i] = 0u; hist2[i] = 0u; }
}

// ---------------- K1: fused edge GEMM -> transition logits (WMMA) ----------------
// Block = 256 threads (8 waves), 32 edges. Wave w owns M-tile (w>>2) and N-tiles (w&3)*4..+3.
// K-outer loop: one A fragment load feeds 4 WMMAs (4 accumulators) -> 4x A reuse, deep ILP.
__launch_bounds__(256)
__global__ void k1_edge_logits(const int* __restrict__ eg_idx, const int* __restrict__ node_i,
                               const int* __restrict__ node_j,
                               const float* __restrict__ mem, const float* __restrict__ rel,
                               const float* __restrict__ qsrc, const float* __restrict__ qrel,
                               const _Float16* __restrict__ Wl_s, const float* __restrict__ bl,
                               const _Float16* __restrict__ Wr_s, const float* __restrict__ br,
                               const _Float16* __restrict__ Wc_s, const float* __restrict__ bc,
                               float* __restrict__ logits, int E)
{
    // A-fragment-ordered staging: [mtile(2)][ktile(16)][lane(32)][e(16)] halves
    __shared__ _Float16 lx[TILE_E * IDD] __attribute__((aligned(32)));   // 32KB
    __shared__ _Float16 rx[TILE_E * IDD] __attribute__((aligned(32)));   // 32KB
    __shared__ float    lbuf[TILE_E * ODD];                              // 32KB, natural [m][n]
    __shared__ _Float16 rp[2 * 8 * 512] __attribute__((aligned(32)));    // 16KB, A-frag for Wc GEMM
    __shared__ float    lacc[TILE_E];
    __shared__ int      sni[TILE_E], snj[TILE_E], seg[TILE_E];

    const int tid = threadIdx.x;
    const int eb  = blockIdx.x * TILE_E;
    const int w   = tid >> 5;   // wave id 0..7
    const int l   = tid & 31;   // lane

    if (tid < TILE_E) {
        int e = eb + tid; if (e >= E) e = E - 1;
        sni[tid] = node_i[e];
        snj[tid] = node_j[e];
        seg[tid] = eg_idx[e];
        lacc[tid] = 0.0f;
    }
    __syncthreads();

    // ---- stage A fragments for left_x / right_x (f32 gather -> f16, fragment order) ----
    for (int idx = tid; idx < TILE_E * IDD; idx += 256) {
        int e    = idx & 15;
        int lane = (idx >> 4) & 31;
        int tile = idx >> 9;           // mt*16 + kt
        int kt = tile & 15, mt = tile >> 4;
        int g = lane >> 4, mr = lane & 15;
        int m = mt * 16 + mr;
        int v = e >> 1, hb = e & 1;
        int kloc = ((v & 4) << 2) + ((v & 3) << 1) + hb + (g << 3); // A layout: g*8
        int K = kt * 32 + kloc;
        int edge = eb + m; if (edge >= E) edge = E - 1;
        float lv, rv;
        if (K < DD) {
            lv = mem[sni[m] * DD + K];
            rv = mem[snj[m] * DD + K];
        } else if (K < 2 * DD) {
            float t = rel[edge * DD + (K - DD)]; lv = t; rv = t;
        } else if (K < 3 * DD) {
            float t = qsrc[seg[m] * DD + (K - 2 * DD)]; lv = t; rv = t;
        } else {
            float t = qrel[seg[m] * DD + (K - 3 * DD)]; lv = t; rv = t;
        }
        lx[idx] = (_Float16)lv;
        rx[idx] = (_Float16)rv;
    }
    __syncthreads();

    const int g   = l >> 4;        // half-wave
    const int ln  = l & 15;        // lane within half
    const int mt  = w >> 2;        // M-tile owned by this wave
    const int ntb = (w & 3) << 2;  // first of 4 N-tiles owned by this wave

    // ---- GEMM1-left: l = leaky(left_x @ Wl + bl)  -> lbuf (f32) ----
    {
        floatx8 acc[4] = {};
#pragma unroll
        for (int kt = 0; kt < IDD / 32; ++kt) {
            half16 a = *reinterpret_cast<const half16*>(&lx[(((mt * 16 + kt) << 5) + l) << 4]);
#pragma unroll
            for (int j = 0; j < 4; ++j) {
                half16 b = *reinterpret_cast<const half16*>(
                    Wl_s + ((((kt << 4) + ntb + j) << 5) + l) * 16);
                acc[j] = __builtin_amdgcn_wmma_f32_16x16x32_f16(false, a, false, b,
                                                                (short)0, acc[j], false, false);
            }
        }
#pragma unroll
        for (int j = 0; j < 4; ++j) {
#pragma unroll
            for (int r = 0; r < 8; ++r) {
                int m = mt * 16 + r + (g << 3);
                int n = (ntb + j) * 16 + ln;
                float x = acc[j][r] + bl[n];
                x = x > 0.0f ? x : 0.01f * x;      // leaky_relu, slope 0.01
                lbuf[m * ODD + n] = x;
            }
        }
    }

    // ---- GEMM1-right: rp = leaky(right_x @ Wr + br)  -> rp (f16 A-frag for Wc GEMM) ----
    {
        floatx8 acc[4] = {};
#pragma unroll
        for (int kt = 0; kt < IDD / 32; ++kt) {
            half16 a = *reinterpret_cast<const half16*>(&rx[(((mt * 16 + kt) << 5) + l) << 4]);
#pragma unroll
            for (int j = 0; j < 4; ++j) {
                half16 b = *reinterpret_cast<const half16*>(
                    Wr_s + ((((kt << 4) + ntb + j) << 5) + l) * 16);
                acc[j] = __builtin_amdgcn_wmma_f32_16x16x32_f16(false, a, false, b,
                                                                (short)0, acc[j], false, false);
            }
        }
#pragma unroll
        for (int j = 0; j < 4; ++j) {
#pragma unroll
            for (int r = 0; r < 8; ++r) {
                int m = mt * 16 + r + (g << 3);
                int n = (ntb + j) * 16 + ln;
                float x = acc[j][r] + br[n];
                x = x > 0.0f ? x : 0.01f * x;
                // scatter into A-fragment layout; K2 = n (output col becomes contraction dim)
                int K2 = n;
                int kt2 = K2 >> 5, k2l = K2 & 31;
                int g2 = (k2l >> 3) & 1;
                int v2 = ((k2l >> 4) << 2) | ((k2l >> 1) & 3);
                int h2 = k2l & 1;
                int lane2 = (m & 15) + (g2 << 4);
                rp[((((m >> 4) * 8 + kt2) << 5) + lane2) * 16 + ((v2 << 1) | h2)] = (_Float16)x;
            }
        }
    }
    __syncthreads();

    // ---- GEMM2: r = rp @ Wc + bc ; logits += sum_n l*r ----
    {
        floatx8 acc[4] = {};
#pragma unroll
        for (int kt = 0; kt < ODD / 32; ++kt) {
            half16 a = *reinterpret_cast<const half16*>(&rp[(((mt * 8 + kt) << 5) + l) << 4]);
#pragma unroll
            for (int j = 0; j < 4; ++j) {
                half16 b = *reinterpret_cast<const half16*>(
                    Wc_s + ((((kt << 4) + ntb + j) << 5) + l) * 16);
                acc[j] = __builtin_amdgcn_wmma_f32_16x16x32_f16(false, a, false, b,
                                                                (short)0, acc[j], false, false);
            }
        }
#pragma unroll
        for (int j = 0; j < 4; ++j) {
#pragma unroll
            for (int r = 0; r < 8; ++r) {
                int m = mt * 16 + r + (g << 3);
                int n = (ntb + j) * 16 + ln;
                float prod = (acc[j][r] + bc[n]) * lbuf[m * ODD + n];
                // reduce across the 16 lanes holding row m (n varies)
                prod += __shfl_xor(prod, 1, 32);
                prod += __shfl_xor(prod, 2, 32);
                prod += __shfl_xor(prod, 4, 32);
                prod += __shfl_xor(prod, 8, 32);
                if (ln == 0) atomicAdd(&lacc[m], prod);
            }
        }
    }
    __syncthreads();

    if (tid < TILE_E) {
        int e = eb + tid;
        if (e < E) logits[e] = lacc[tid];
    }
}

// ---------------- monotone float<->uint key (total order over floats) ----------------
__device__ __forceinline__ unsigned fkey(float f) {
    unsigned u = __float_as_uint(f);
    return (u & 0x80000000u) ? ~u : (u | 0x80000000u);
}
__device__ __forceinline__ float fdec(unsigned k) {
    return (k & 0x80000000u) ? __uint_as_float(k ^ 0x80000000u) : __uint_as_float(~k);
}

// K2: segment max over node_i
__global__ void k2_segmax(const float* __restrict__ logits, const int* __restrict__ node_i,
                          unsigned* __restrict__ maxkey, int E)
{
    int e = blockIdx.x * blockDim.x + threadIdx.x;
    if (e >= E) return;
    atomicMax(&maxkey[node_i[e]], fkey(logits[e]));
}

// K3: exp(logit - max) and segment sum
__global__ void k3_expsum(const float* __restrict__ logits, const int* __restrict__ node_i,
                          const unsigned* __restrict__ maxkey, float* __restrict__ soft,
                          float* __restrict__ sumexp, int E)
{
    int e = blockIdx.x * blockDim.x + threadIdx.x;
    if (e >= E) return;
    int ni = node_i[e];
    float v = __expf(logits[e] - fdec(maxkey[ni]));
    soft[e] = v;
    atomicAdd(&sumexp[ni], v);
}

// K4: normalize softmax, compute target_att, histogram pass 1 (top byte of float bits)
__global__ void k4_norm_hist(float* __restrict__ soft, const float* __restrict__ sumexp,
                             const int* __restrict__ node_i, const float* __restrict__ node_score,
                             const int* __restrict__ eg_idx, float* __restrict__ tatt,
                             unsigned* __restrict__ hist1, int E)
{
    int e = blockIdx.x * blockDim.x + threadIdx.x;
    if (e >= E) return;
    int ni = node_i[e];
    float s = soft[e] / sumexp[ni];
    soft[e] = s;
    float t = s * node_score[ni];            // >= 0
    tatt[e] = t;
    unsigned bin = __float_as_uint(t) >> 24;
    atomicAdd(&hist1[eg_idx[e] * 256 + bin], 1u);
}

// K5: per-group top-k select level 1 (top byte)
__global__ void k5_sel1(const unsigned* __restrict__ hist1, const int* __restrict__ kptr,
                        int* __restrict__ b1, unsigned* __restrict__ cnt, int B)
{
    int gq = blockIdx.x * blockDim.x + threadIdx.x;
    if (gq >= B) return;
    unsigned k = (unsigned)(*kptr);
    unsigned cum = 0; int bb = -1;
    for (int b = 255; b >= 0; --b) {
        unsigned c = hist1[gq * 256 + b];
        if (cum + c >= k) { bb = b; break; }
        cum += c;
    }
    b1[gq] = bb;          // -1 => group has < k edges, keep all
    cnt[gq] = cum;        // count strictly above boundary bin
}

// K6: histogram pass 2 (second byte) within boundary bin
__global__ void k6_hist2(const float* __restrict__ tatt, const int* __restrict__ eg_idx,
                         const int* __restrict__ b1, unsigned* __restrict__ hist2, int E)
{
    int e = blockIdx.x * blockDim.x + threadIdx.x;
    if (e >= E) return;
    int gq = eg_idx[e];
    int bb = b1[gq];
    if (bb < 0) return;
    unsigned bits = __float_as_uint(tatt[e]);
    if ((int)(bits >> 24) == bb)
        atomicAdd(&hist2[gq * 256 + ((bits >> 16) & 255u)], 1u);
}

// K7: select level 2 -> 16-bit threshold key
__global__ void k7_sel2(const unsigned* __restrict__ hist2, const int* __restrict__ kptr,
                        const int* __restrict__ b1, const unsigned* __restrict__ cnt,
                        unsigned* __restrict__ thr, int B)
{
    int gq = blockIdx.x * blockDim.x + threadIdx.x;
    if (gq >= B) return;
    int bb = b1[gq];
    if (bb < 0) { thr[gq] = 0u; return; }
    unsigned k = (unsigned)(*kptr);
    unsigned cum = cnt[gq];
    unsigned t = ((unsigned)bb << 24);
    for (int b = 255; b >= 0; --b) {
        unsigned c = hist2[gq * 256 + b];
        if (cum + c >= k) { t = ((unsigned)bb << 24) | ((unsigned)b << 16); break; }
        cum += c;
    }
    thr[gq] = t;
}

// K8: keep-masked scatter-add into node_j
__global__ void k8_scatter(const float* __restrict__ soft, const float* __restrict__ tatt,
                           const int* __restrict__ eg_idx, const int* __restrict__ node_j,
                           const unsigned* __restrict__ thr, float* __restrict__ out, int E)
{
    int e = blockIdx.x * blockDim.x + threadIdx.x;
    if (e >= E) return;
    float t = tatt[e];
    if (__float_as_uint(t) >= thr[eg_idx[e]])
        atomicAdd(&out[node_j[e]], soft[e] * t);
}

extern "C" void kernel_launch(void* const* d_in, const int* in_sizes, int n_in,
                              void* d_out, int out_size, void* d_ws, size_t ws_size,
                              hipStream_t stream)
{
    const int*   eg_idx = (const int*)  d_in[0];
    const int*   node_i = (const int*)  d_in[1];
    const int*   node_j = (const int*)  d_in[2];
    const float* nscore = (const float*)d_in[3];
    const float* mem    = (const float*)d_in[4];
    const float* rel    = (const float*)d_in[5];
    const float* qsrc   = (const float*)d_in[6];
    const float* qrel   = (const float*)d_in[7];
    const float* Wl     = (const float*)d_in[8];
    const float* bl     = (const float*)d_in[9];
    const float* Wr     = (const float*)d_in[10];
    const float* br     = (const float*)d_in[11];
    const float* Wc     = (const float*)d_in[12];
    const float* bc     = (const float*)d_in[13];
    const int*   kptr   = (const int*)  d_in[14];

    const int E = in_sizes[0];
    const int N = in_sizes[3];
    const int Drt = in_sizes[4] / N;           // = 128
    const int B = in_sizes[6] / Drt;           // = 64
    float* out = (float*)d_out;

    // ---- workspace carve-up ----
    char* ws = (char*)d_ws;
    size_t off = 0;
    auto carve = [&](size_t bytes) { size_t r = off; off = (off + bytes + 255) & ~(size_t)255; return r; };
    _Float16* Wl_s  = (_Float16*)(ws + carve((size_t)IDD * ODD * 2));
    _Float16* Wr_s  = (_Float16*)(ws + carve((size_t)IDD * ODD * 2));
    _Float16* Wc_s  = (_Float16*)(ws + carve((size_t)ODD * ODD * 2));
    float*    logits= (float*)   (ws + carve((size_t)E * 4));
    float*    soft  = (float*)   (ws + carve((size_t)E * 4));
    float*    tatt  = (float*)   (ws + carve((size_t)E * 4));
    unsigned* maxkey= (unsigned*)(ws + carve((size_t)N * 4));
    float*    sumexp= (float*)   (ws + carve((size_t)N * 4));
    unsigned* hist1 = (unsigned*)(ws + carve((size_t)B * 256 * 4));
    unsigned* hist2 = (unsigned*)(ws + carve((size_t)B * 256 * 4));
    int*      b1    = (int*)     (ws + carve((size_t)B * 4));
    unsigned* cnt   = (unsigned*)(ws + carve((size_t)B * 4));
    unsigned* thr   = (unsigned*)(ws + carve((size_t)B * 4));
    (void)ws_size; (void)n_in; (void)out_size;

    const int TB = 256;
    const int gE = (E + TB - 1) / TB;
    const int gB = (B + TB - 1) / TB;

    k0_prep<<<1024, TB, 0, stream>>>(Wl, Wr, Wc, Wl_s, Wr_s, Wc_s,
                                     maxkey, sumexp, out, hist1, hist2, N, B);
    k1_edge_logits<<<(E + TILE_E - 1) / TILE_E, TB, 0, stream>>>(
        eg_idx, node_i, node_j, mem, rel, qsrc, qrel,
        Wl_s, bl, Wr_s, br, Wc_s, bc, logits, E);
    k2_segmax<<<gE, TB, 0, stream>>>(logits, node_i, maxkey, E);
    k3_expsum<<<gE, TB, 0, stream>>>(logits, node_i, maxkey, soft, sumexp, E);
    k4_norm_hist<<<gE, TB, 0, stream>>>(soft, sumexp, node_i, nscore, eg_idx, tatt, hist1, E);
    k5_sel1<<<gB, TB, 0, stream>>>(hist1, kptr, b1, cnt, B);
    k6_hist2<<<gE, TB, 0, stream>>>(tatt, eg_idx, b1, hist2, E);
    k7_sel2<<<gB, TB, 0, stream>>>(hist2, kptr, b1, cnt, thr, B);
    k8_scatter<<<gE, TB, 0, stream>>>(soft, tatt, eg_idx, node_j, thr, out, E);
}
